// MultiHeadAttention_22076131901748
// MI455X (gfx1250) — matrix operand order repaired and checked by reading, not compiled, not run on hardware
//
#include <hip/hip_runtime.h>
#include <hip/hip_bf16.h>

#define B_NUM 4
#define S_LEN 2048
#define D_DIM 1024
#define H_NUM 16
#define DK    64

typedef __attribute__((ext_vector_type(16))) __bf16 v16bf;
typedef __attribute__((ext_vector_type(8)))  __bf16 v8bf;
typedef __attribute__((ext_vector_type(8)))  float  v8f;

__device__ __forceinline__ unsigned short f2bf(float f) {
  unsigned int u = __builtin_bit_cast(unsigned int, f);
  u += 0x7fffu + ((u >> 16) & 1u);         // round-to-nearest-even
  return (unsigned short)(u >> 16);
}

__device__ __forceinline__ v8f zero_v8f() {
  v8f z;
#pragma unroll
  for (int i = 0; i < 8; ++i) z[i] = 0.0f;
  return z;
}

__device__ __forceinline__ v16bf concat8(v8bf lo, v8bf hi) {
  v16bf r;
#pragma unroll
  for (int i = 0; i < 8; ++i) { r[i] = lo[i]; r[i + 8] = hi[i]; }
  return r;
}

// A fragment (16x32 bf16, MxK) from a row-major tile row (rowPtr=&tile[m][0], m=lane%16).
// j selects the 32-wide K chunk. e0..7 -> k=j*32+half*8+e ; e8..15 -> k=j*32+16+half*8+(e-8)
__device__ __forceinline__ v16bf load_a_frag(const unsigned short* rowPtr, int j, int hf) {
  const char* p = (const char*)rowPtr + j * 64 + hf * 16;
  v8bf lo = *(const v8bf*)p;
  v8bf hi = *(const v8bf*)(p + 32);
  return concat8(lo, hi);
}

// B fragment (32x16 bf16, KxN) from an [N][K] (transposed) tile row (rowPtr=&tileT[n][0]).
// e0..7 -> k=j*32+half*8+e ; e8..15 -> k=j*32+16+half*8+(e-8)
__device__ __forceinline__ v16bf load_b_frag(const unsigned short* rowPtr, int j, int hf) {
  const char* p = (const char*)rowPtr + j * 64 + hf * 16;
  v8bf lo = *(const v8bf*)p;
  v8bf hi = *(const v8bf*)(p + 32);
  return concat8(lo, hi);
}

__device__ __forceinline__ v8f wmma_bf16(v16bf a, v16bf b, v8f c) {
  return __builtin_amdgcn_wmma_f32_16x16x32_bf16(false, a, false, b, (short)0, c,
                                                 false, false);
}

// ---- CDNA5 async memory->LDS copy (ASYNCcnt-tracked, no VGPR data) -------
__device__ __forceinline__ unsigned lds_off(const void* p) {
  return (unsigned)(unsigned long long)p;   // low 32 bits of flat LDS addr = LDS offset
}

__device__ __forceinline__ void async_ld16(unsigned dst_lds, const void* src) {
  asm volatile("global_load_async_to_lds_b128 %0, %1, off"
               :: "v"(dst_lds), "v"(src) : "memory");
}

__device__ __forceinline__ void wait_async_all() {
  asm volatile("s_wait_asynccnt 0x0" ::: "memory");
}

// ---------------------------------------------------------------------------
// Weight transpose+convert: Wt[n][k] bf16 = W[k][n] f32.  32x32 LDS tiles.
// ---------------------------------------------------------------------------
__global__ __launch_bounds__(256)
void transpose_convert_w(const float* __restrict__ W, unsigned short* __restrict__ Wt) {
  __shared__ unsigned short tile[32][33];
  const int kb = blockIdx.x * 32;
  const int nb = blockIdx.y * 32;
  const int tk = threadIdx.x >> 3;          // 0..31
  const int tn = (threadIdx.x & 7) * 4;     // 0..28
  const float4 f = *(const float4*)(W + (size_t)(kb + tk) * D_DIM + nb + tn);
  tile[tn + 0][tk] = f2bf(f.x);
  tile[tn + 1][tk] = f2bf(f.y);
  tile[tn + 2][tk] = f2bf(f.z);
  tile[tn + 3][tk] = f2bf(f.w);
  __syncthreads();
  const int wn = threadIdx.x >> 3;
  const int wk = (threadIdx.x & 7) * 4;
  unsigned short* dst = Wt + (size_t)(nb + wn) * D_DIM + kb + wk;
  unsigned int lo = (unsigned int)tile[wn][wk + 0] | ((unsigned int)tile[wn][wk + 1] << 16);
  unsigned int hi = (unsigned int)tile[wn][wk + 2] | ((unsigned int)tile[wn][wk + 3] << 16);
  *(uint2*)dst = make_uint2(lo, hi);
}

// ---------------------------------------------------------------------------
// GEMM: out[M,1024] = A[M,1024] @ W + bias.  Wt is bf16 [n][k] (B-frag ready),
// staged by async LDS copies. BK=64 -> 16 WMMAs per stage.
// mode 0: out bf16 [b,h,s,dk] (Q,K)   mode 1: out bf16 [b,h,dk,s] (V)
// mode 2: A bf16, out f32 [M,1024] (final projection)
// ---------------------------------------------------------------------------
__global__ __launch_bounds__(256)
void gemm_bias_kernel(const void* __restrict__ Aptr, int a_is_bf16,
                      const unsigned short* __restrict__ Wt,
                      const float* __restrict__ bias,
                      void* __restrict__ out, int mode) {
  __shared__ __align__(16) unsigned short As[128][64];   // [m][k]  16 KB
  __shared__ __align__(16) unsigned short Bs[128][64];   // [n][k]  16 KB

  const int tid  = threadIdx.x;
  const int lane = tid & 31;
  const int wv   = tid >> 5;
  const int hf   = (lane >> 4) & 1;
  const int lc   = lane & 15;

  const int rowBase = blockIdx.y * 128;
  const int colBase = blockIdx.x * 128;
  const int waveM = wv >> 1;          // 0..3
  const int waveN = wv & 1;           // 0..1

  v8f acc[2][4];
#pragma unroll
  for (int t = 0; t < 2; ++t)
#pragma unroll
    for (int u = 0; u < 4; ++u) acc[t][u] = zero_v8f();

  const float* Af = (const float*)Aptr;
  const unsigned short* Ab = (const unsigned short*)Aptr;

  for (int ks = 0; ks < D_DIM / 64; ++ks) {
    const int k0 = ks * 64;

    // ---- async-stage B tile (and A tile when bf16): contiguous b128 copies ----
#pragma unroll
    for (int p = 0; p < 2; ++p) {
      const int idx = tid + p * 256;
      const int r = idx >> 2;                 // 0..127
      const int c = (idx & 3) * 16;           // 0,16,32,48
      const unsigned short* bsrc = Wt + (size_t)(colBase + r) * D_DIM + k0 + c;
      async_ld16(lds_off(&Bs[r][c]),     bsrc);
      async_ld16(lds_off(&Bs[r][c + 8]), bsrc + 8);
      if (a_is_bf16) {
        const unsigned short* asrc = Ab + (size_t)(rowBase + r) * D_DIM + k0 + c;
        async_ld16(lds_off(&As[r][c]),     asrc);
        async_ld16(lds_off(&As[r][c + 8]), asrc + 8);
      }
    }
    // ---- sync-stage A tile with f32->bf16 conversion (overlaps async B) ----
    if (!a_is_bf16) {
      const int r0 = tid >> 4;                // 0..15
      const int c0 = (tid & 15) * 4;          // 0..60
#pragma unroll
      for (int p = 0; p < 8; ++p) {
        const int r = r0 + p * 16;
        const float4 f = *(const float4*)(Af + (size_t)(rowBase + r) * D_DIM + k0 + c0);
        unsigned int lo  = (unsigned int)f2bf(f.x) | ((unsigned int)f2bf(f.y) << 16);
        unsigned int hi2 = (unsigned int)f2bf(f.z) | ((unsigned int)f2bf(f.w) << 16);
        *(uint2*)&As[r][c0] = make_uint2(lo, hi2);
      }
    }
    wait_async_all();
    __syncthreads();

#pragma unroll
    for (int j = 0; j < 2; ++j) {
#pragma unroll
      for (int t = 0; t < 2; ++t) {
        const int mrow = waveM * 32 + t * 16 + lc;
        v16bf a = load_a_frag(&As[mrow][0], j, hf);
#pragma unroll
        for (int u = 0; u < 4; ++u) {
          const int ncol = waveN * 64 + u * 16 + lc;
          v16bf b = load_b_frag(&Bs[ncol][0], j, hf);
          acc[t][u] = wmma_bf16(a, b, acc[t][u]);
        }
      }
    }
    __syncthreads();
  }

  // ---- epilogue: bias + layout-specific store ----
#pragma unroll
  for (int t = 0; t < 2; ++t) {
#pragma unroll
    for (int u = 0; u < 4; ++u) {
#pragma unroll
      for (int v = 0; v < 8; ++v) {
        const int grow = rowBase + waveM * 32 + t * 16 + v + hf * 8;
        const int gcol = colBase + waveN * 64 + u * 16 + lc;
        const float val = acc[t][u][v] + bias[gcol];
        if (mode == 2) {
          ((float*)out)[(size_t)grow * D_DIM + gcol] = val;
        } else {
          const int b  = grow >> 11;
          const int s  = grow & (S_LEN - 1);
          const int h  = gcol >> 6;
          const int dk = gcol & (DK - 1);
          size_t off;
          if (mode == 0)
            off = (((size_t)b * H_NUM + h) * S_LEN + s) * DK + dk;   // [b,h,s,dk]
          else
            off = (((size_t)b * H_NUM + h) * DK + dk) * S_LEN + s;   // [b,h,dk,s]
          ((unsigned short*)out)[off] = f2bf(val);
        }
      }
    }
  }
}

// ---------------------------------------------------------------------------
// Flash attention, double-buffered async K/V staging.
// Block = 128 q-rows of one (b,h); 8 waves x 16 rows; kv tiles of 64.
// ---------------------------------------------------------------------------
__device__ __forceinline__ void stage_kv_async(const unsigned short* Kh,
                                               const unsigned short* Vh,
                                               int kv,
                                               unsigned short (*Ksb)[64],
                                               unsigned short (*Vsb)[64],
                                               int tid) {
  const int r = tid >> 2;                 // 0..63
  const int c = (tid & 3) * 16;
  const unsigned short* kg = Kh + (size_t)(kv + r) * DK + c;
  const unsigned short* vg = Vh + (size_t)r * S_LEN + kv + c;
  async_ld16(lds_off(&Ksb[r][c]),     kg);
  async_ld16(lds_off(&Ksb[r][c + 8]), kg + 8);
  async_ld16(lds_off(&Vsb[r][c]),     vg);
  async_ld16(lds_off(&Vsb[r][c + 8]), vg + 8);
}

__global__ __launch_bounds__(256)
void flash_attn_kernel(const unsigned short* __restrict__ Q,
                       const unsigned short* __restrict__ K,
                       const unsigned short* __restrict__ Vt,
                       unsigned short* __restrict__ ctx) {
  __shared__ __align__(16) unsigned short Ks[2][64][64];   // [kv][dk]  16 KB
  __shared__ __align__(16) unsigned short Vs[2][64][64];   // [dk][kv]  16 KB
  __shared__ __align__(16) unsigned short Ps[8][16][64];   // per-wave P, 16 KB

  const int tid  = threadIdx.x;
  const int lane = tid & 31;
  const int wv   = tid >> 5;
  const int hf   = (lane >> 4) & 1;
  const int lc   = lane & 15;

  const int bh    = blockIdx.y;                 // b*16 + h
  const int qBase = blockIdx.x * 128 + wv * 16;

  const unsigned short* Qh = Q  + (size_t)bh * S_LEN * DK;
  const unsigned short* Kh = K  + (size_t)bh * S_LEN * DK;
  const unsigned short* Vh = Vt + (size_t)bh * DK * S_LEN;

  v16bf qf[2];
#pragma unroll
  for (int j = 0; j < 2; ++j)
    qf[j] = load_a_frag(Qh + (size_t)(qBase + lc) * DK, j, hf);

  v8f oacc[4];
#pragma unroll
  for (int u = 0; u < 4; ++u) oacc[u] = zero_v8f();
  float mrow[8], lrow[8];
#pragma unroll
  for (int v = 0; v < 8; ++v) { mrow[v] = -1e30f; lrow[v] = 0.0f; }

  // prologue: stage tile 0
  stage_kv_async(Kh, Vh, 0, Ks[0], Vs[0], tid);

  const int nIter = S_LEN / 64;
  for (int it = 0; it < nIter; ++it) {
    const int buf = it & 1;
    wait_async_all();            // own async copies done (tile `buf` staged by me)
    __syncthreads();             // everyone's copies visible; prev compute finished
    if (it + 1 < nIter)          // overlap next tile's DMA with this tile's math
      stage_kv_async(Kh, Vh, (it + 1) * 64, Ks[buf ^ 1], Vs[buf ^ 1], tid);

    // ---- scores: S = Q . K^T (16 x 64 per wave) ----
    v8f sacc[4];
#pragma unroll
    for (int u = 0; u < 4; ++u) sacc[u] = zero_v8f();
#pragma unroll
    for (int j = 0; j < 2; ++j) {
#pragma unroll
      for (int u = 0; u < 4; ++u) {
        v16bf b = load_b_frag(&Ks[buf][u * 16 + lc][0], j, hf);
        sacc[u] = wmma_bf16(qf[j], b, sacc[u]);
      }
    }

    // ---- online softmax (row r = v + 8*hf, cols striped over 16 lanes) ----
#pragma unroll
    for (int v = 0; v < 8; ++v) {
      float mx = -1e30f;
#pragma unroll
      for (int u = 0; u < 4; ++u) {
        const float s = sacc[u][v] * 0.125f;    // 1/sqrt(64)
        sacc[u][v] = s;
        mx = fmaxf(mx, s);
      }
#pragma unroll
      for (int i = 1; i < 16; i <<= 1) mx = fmaxf(mx, __shfl_xor(mx, i, 32));
      const float mnew = fmaxf(mrow[v], mx);
      const float corr = __expf(mrow[v] - mnew);
      float rs = 0.0f;
#pragma unroll
      for (int u = 0; u < 4; ++u) {
        const float p = __expf(sacc[u][v] - mnew);
        sacc[u][v] = p;
        rs += p;
      }
#pragma unroll
      for (int i = 1; i < 16; i <<= 1) rs += __shfl_xor(rs, i, 32);
      lrow[v] = lrow[v] * corr + rs;
      mrow[v] = mnew;
#pragma unroll
      for (int w = 0; w < 4; ++w) oacc[w][v] *= corr;
    }

    // ---- P: C-layout -> A-layout via wave-private LDS ----
#pragma unroll
    for (int u = 0; u < 4; ++u)
#pragma unroll
      for (int v = 0; v < 8; ++v)
        Ps[wv][v + 8 * hf][u * 16 + lc] = f2bf(sacc[u][v]);

    // ---- out += P . V ----
#pragma unroll
    for (int j = 0; j < 2; ++j) {
      v16bf a = load_a_frag(&Ps[wv][lc][0], j, hf);
#pragma unroll
      for (int w = 0; w < 4; ++w) {
        v16bf b = load_b_frag(&Vs[buf][w * 16 + lc][0], j, hf);
        oacc[w] = wmma_bf16(a, b, oacc[w]);
      }
    }
  }

  // ---- epilogue: normalize, store ctx bf16 [b,s,h*64+dk] ----
  const int b = bh >> 4, h = bh & 15;
#pragma unroll
  for (int v = 0; v < 8; ++v) {
    const float inv = 1.0f / lrow[v];
    const int s = qBase + v + 8 * hf;
#pragma unroll
    for (int w = 0; w < 4; ++w) {
      const int col = h * DK + w * 16 + lc;
      ctx[((size_t)b * S_LEN + s) * D_DIM + col] = f2bf(oacc[w][v] * inv);
    }
  }
}

extern "C" void kernel_launch(void* const* d_in, const int* in_sizes, int n_in,
                              void* d_out, int out_size, void* d_ws, size_t ws_size,
                              hipStream_t stream) {
  (void)in_sizes; (void)n_in; (void)out_size; (void)ws_size;
  const float* query = (const float*)d_in[0];
  const float* key   = (const float*)d_in[1];
  const float* value = (const float*)d_in[2];
  const float* Wq = (const float*)d_in[3];
  const float* bq = (const float*)d_in[4];
  const float* Wk = (const float*)d_in[5];
  const float* bk = (const float*)d_in[6];
  const float* Wv = (const float*)d_in[7];
  const float* bv = (const float*)d_in[8];
  const float* Wo = (const float*)d_in[9];
  const float* bo = (const float*)d_in[10];

  const size_t headElems = (size_t)B_NUM * H_NUM * S_LEN * DK;   // 8M bf16 each
  const size_t wElems    = (size_t)D_DIM * D_DIM;                // 1M bf16 each
  unsigned short* Qws  = (unsigned short*)d_ws;
  unsigned short* Kws  = Qws  + headElems;
  unsigned short* Vtws = Kws  + headElems;
  unsigned short* Cws  = Vtws + headElems;
  unsigned short* WqT  = Cws  + headElems;
  unsigned short* WkT  = WqT  + wElems;
  unsigned short* WvT  = WkT  + wElems;
  unsigned short* WoT  = WvT  + wElems;

  dim3 gT(D_DIM / 32, D_DIM / 32);                  // (32, 32)
  transpose_convert_w<<<gT, 256, 0, stream>>>(Wq, WqT);
  transpose_convert_w<<<gT, 256, 0, stream>>>(Wk, WkT);
  transpose_convert_w<<<gT, 256, 0, stream>>>(Wv, WvT);
  transpose_convert_w<<<gT, 256, 0, stream>>>(Wo, WoT);

  dim3 gGemm(D_DIM / 128, (B_NUM * S_LEN) / 128);   // (8, 64)
  gemm_bias_kernel<<<gGemm, 256, 0, stream>>>(query, 0, WqT, bq, Qws, 0);
  gemm_bias_kernel<<<gGemm, 256, 0, stream>>>(key,   0, WkT, bk, Kws, 0);
  gemm_bias_kernel<<<gGemm, 256, 0, stream>>>(value, 0, WvT, bv, Vtws, 1);

  dim3 gAttn(S_LEN / 128, B_NUM * H_NUM);           // (16, 64)
  flash_attn_kernel<<<gAttn, 256, 0, stream>>>(Qws, Kws, Vtws, Cws);

  gemm_bias_kernel<<<gGemm, 256, 0, stream>>>(Cws, 1, WoT, bo, d_out, 2);
}
